// GATExtractPart_18176301596820
// MI455X (gfx1250) — matrix operand order repaired and checked by reading, not compiled, not run on hardware
//
#include <hip/hip_runtime.h>
#include <math.h>

typedef __attribute__((ext_vector_type(16))) __bf16 v16bf;
typedef __attribute__((ext_vector_type(8)))  float  v8f;

#define NEG_SLOPE 0.2f
#define LN_EPS 1e-5f

// ---------------------------------------------------------------- utilities

__device__ inline void atomicMaxF(float* addr, float val) {
  unsigned int* ua = (unsigned int*)addr;
  unsigned int old = *ua;
  while (__uint_as_float(old) < val) {
    unsigned int assumed = old;
    old = atomicCAS(ua, assumed, __float_as_uint(val));
    if (old == assumed) break;
  }
}

__global__ void fill_k(float* p, float v, int n) {
  int i = blockIdx.x * blockDim.x + threadIdx.x;
  if (i < n) p[i] = v;
}

// ------------------------------------------------- self-loop edge_attr mean

__global__ void sl_accum_k(const int* __restrict__ ei, const float* __restrict__ ea,
                           float* sums, float* cnt, int E) {
  int e = blockIdx.x * blockDim.x + threadIdx.x;
  if (e >= E) return;
  int dst = ei[E + e];
#pragma unroll
  for (int k = 0; k < 6; k++) atomicAdd(&sums[dst * 6 + k], ea[(size_t)e * 6 + k]);
  atomicAdd(&cnt[dst], 1.0f);
}

__global__ void sl_final_k(float* sums, const float* __restrict__ cnt, int Nn) {
  int i = blockIdx.x * blockDim.x + threadIdx.x;
  if (i >= Nn * 6) return;
  sums[i] = sums[i] / fmaxf(cnt[i / 6], 1.0f);
}

// --------------------------------------------------------- bf16 WMMA GEMM
// C[M,N] = A[M,K] @ B[K,N], fp32 in/out, v_wmma_f32_16x16x32_bf16 inner.
// 256 threads = 8 waves; block computes a 64x64 C tile; wave w owns two
// 16x16 tiles (mb = w>>1, nb = (w&1)*2 and nb+1) -> A-fragment reuse, 2 WMMA
// per wave per k-step, 16 WMMA per block per barrier pair.
// LDS tiles are stored directly in CDNA5 WMMA fragment order so the per-lane
// fragment load is one contiguous 32B (ds_load_b128 x2).
// K loop is manually split: predicate-free main loop over K&~31, plus one
// guarded tail step (only taken when K % 32 != 0).
// N % 64 == 0; M arbitrary (uniform row clamp + store guard).

__device__ __forceinline__ void wmma_step(
    const v16bf (&sAf)[4][32], const v16bf (&sBf)[4][32],
    int mb, int nb, int lane, v8f& acc0, v8f& acc1) {
  v16bf af = sAf[mb][lane];
  v16bf b0 = sBf[nb][lane];
  v16bf b1 = sBf[nb + 1][lane];
  acc0 = __builtin_amdgcn_wmma_f32_16x16x32_bf16(
      false, af, false, b0, (short)0, acc0, false, false);
  acc1 = __builtin_amdgcn_wmma_f32_16x16x32_bf16(
      false, af, false, b1, (short)0, acc1, false, false);
}

__global__ __launch_bounds__(256) void gemm_bf16_wmma_k(
    const float* __restrict__ A, const float* __restrict__ Bm,
    float* __restrict__ C, int M, int N, int K) {
  __shared__ v16bf sAf[4][32];   // A 64x32 tile, fragment layout [rowblock][lane]
  __shared__ v16bf sBf[4][32];   // B 32x64 tile, fragment layout [colblock][lane]

  const int t = threadIdx.x;
  const int w = t >> 5, lane = t & 31;
  const int mBase = blockIdx.y * 64;
  const int nBase = blockIdx.x * 64;
  const int mb = w >> 1;          // wave's row block (0..3)
  const int nb = (w & 1) * 2;     // wave's first col block (0 or 2)

  // per-thread staging coordinates (8 contiguous elements each)
  const int abase = t * 8;
  const int ar  = abase >> 5;            // A tile row 0..63
  const int ac0 = abase & 31;            // A tile col 0,8,16,24
  const int arg = mBase + ar;
  const bool arow_ok = arg < M;
  const float* aRow = A + (size_t)(arow_ok ? arg : M - 1) * K;
  __bf16* aDst = (__bf16*)&sAf[ar >> 4][0];

  const int bkk = abase >> 6;            // B tile k row 0..31
  const int bn0 = abase & 63;            // 0,8,...,56

  // precompute LDS half-indices (data-independent)
  int aIdx[8], bIdx[8];
#pragma unroll
  for (int i = 0; i < 8; i++) {
    int c = ac0 + i;
    int hh = (c >> 3) & 1;
    int vv = ((c >> 4) & 1) * 4 + ((c & 7) >> 1);
    aIdx[i] = ((ar & 15) + 16 * hh) * 16 + vv * 2 + (c & 1);
    int n = bn0 + i;
    bIdx[i] = ((n & 15) + 16 * ((bkk >> 3) & 1)) * 16 + ((bkk >> 4) & 1) * 8 + (bkk & 7);
  }
  __bf16* bDst = (__bf16*)&sBf[bn0 >> 4][0];

  v8f acc0 = {}, acc1 = {};
  const int kfull = K & ~31;

  // ---------------- predicate-free main loop ----------------
  for (int k0 = 0; k0 < kfull; k0 += 32) {
    {
      const float* ap = aRow + k0;
#pragma unroll
      for (int i = 0; i < 8; i++) {
        float v = ap[ac0 + i];
        v = arow_ok ? v : 0.0f;          // uniform select, hoistable
        aDst[aIdx[i]] = (__bf16)v;
      }
    }
    {
      const float4* bp = (const float4*)(Bm + (size_t)(k0 + bkk) * N + nBase + bn0);
      float4 q0 = bp[0], q1 = bp[1];
      bDst[bIdx[0]] = (__bf16)q0.x; bDst[bIdx[1]] = (__bf16)q0.y;
      bDst[bIdx[2]] = (__bf16)q0.z; bDst[bIdx[3]] = (__bf16)q0.w;
      bDst[bIdx[4]] = (__bf16)q1.x; bDst[bIdx[5]] = (__bf16)q1.y;
      bDst[bIdx[6]] = (__bf16)q1.z; bDst[bIdx[7]] = (__bf16)q1.w;
    }
    __syncthreads();
    wmma_step(sAf, sBf, mb, nb, lane, acc0, acc1);
    __syncthreads();
  }

  // ---------------- guarded tail step (K % 32 != 0) ----------------
  if (kfull < K) {
    {
#pragma unroll
      for (int i = 0; i < 8; i++) {
        int kc = kfull + ac0 + i;
        float v = aRow[kc < K ? kc : 0];           // clamped load
        v = (kc < K && arow_ok) ? v : 0.0f;        // select-zero
        aDst[aIdx[i]] = (__bf16)v;
      }
    }
    {
      int kc = kfull + bkk;
      const float* bp = Bm + (size_t)(kc < K ? kc : 0) * N + nBase + bn0;
      bool kok = kc < K;
#pragma unroll
      for (int i = 0; i < 8; i++) {
        float v = bp[i];
        bDst[bIdx[i]] = (__bf16)(kok ? v : 0.0f);
      }
    }
    __syncthreads();
    wmma_step(sAf, sBf, mb, nb, lane, acc0, acc1);
    __syncthreads();
  }

  // ---- store: C layout row = r + 8*(lane>>4), col = lane&15
  const int hh = lane >> 4, nl = lane & 15;
  const int rowb = mBase + mb * 16 + 8 * hh;
#pragma unroll
  for (int r = 0; r < 8; r++) {
    int row = rowb + r;
    if (row < M) {
      float* cp = C + (size_t)row * N + nBase;
      cp[nb * 16 + nl]       = acc0[r];
      cp[(nb + 1) * 16 + nl] = acc1[r];
    }
  }
}

// ------------------------------------------ folded edge-alpha weight [6,H]

__global__ void weal_k(const float* __restrict__ We, const float* __restrict__ ae,
                       float* out, int H, int C) {
  int t = threadIdx.x;
  if (t >= 6 * H) return;
  int k = t / H, h = t % H;
  float s = 0.0f;
  for (int c = 0; c < C; c++) s += We[(size_t)k * H * C + h * C + c] * ae[h * C + c];
  out[k * H + h] = s;
}

// -------------------------------------------------- per-node alpha factors

__global__ void node_alpha_k(const float* __restrict__ hmat,
                             const float* __restrict__ as_, const float* __restrict__ ad_,
                             float* als, float* ald, int Nn, int H, int C) {
  int i = blockIdx.x * blockDim.x + threadIdx.x;
  if (i >= Nn * H) return;
  int n = i / H, h = i % H;
  const float* hp = hmat + (size_t)n * H * C + (size_t)h * C;
  float s = 0.0f, d = 0.0f;
  for (int c = 0; c < C; c++) {
    float v = hp[c];
    s += v * as_[h * C + c];
    d += v * ad_[h * C + c];
  }
  als[i] = s;
  ald[i] = d;
}

// ------------------------------------------------------ edge attention ops

__global__ void edge_alpha_k(const int* __restrict__ ei, const float* __restrict__ ea,
                             const float* __restrict__ loop_attr, const float* __restrict__ wea,
                             const float* __restrict__ als, const float* __restrict__ ald,
                             float* abuf, float* amax, int E, int Nn, int H) {
  int i = blockIdx.x * blockDim.x + threadIdx.x;
  int Etot = E + Nn;
  if (i >= Etot * H) return;
  int e = i / H, h = i % H;
  int src, dst;
  const float* ep;
  if (e < E) { src = ei[e]; dst = ei[E + e]; ep = ea + (size_t)e * 6; }
  else       { src = dst = e - E;            ep = loop_attr + (size_t)(e - E) * 6; }
  float al = als[src * H + h] + ald[dst * H + h];
#pragma unroll
  for (int k = 0; k < 6; k++) al += ep[k] * wea[k * H + h];
  al = (al > 0.0f) ? al : NEG_SLOPE * al;   // leaky_relu
  abuf[i] = al;
  atomicMaxF(&amax[dst * H + h], al);
}

__global__ void edge_exp_k(const int* __restrict__ ei, const float* __restrict__ amax,
                           float* abuf, float* den, int E, int Nn, int H) {
  int i = blockIdx.x * blockDim.x + threadIdx.x;
  int Etot = E + Nn;
  if (i >= Etot * H) return;
  int e = i / H, h = i % H;
  int dst = (e < E) ? ei[E + e] : (e - E);
  float ex = expf(abuf[i] - amax[dst * H + h]);
  abuf[i] = ex;
  atomicAdd(&den[dst * H + h], ex);
}

__global__ void aggregate_k(const int* __restrict__ ei, const float* __restrict__ hmat,
                            const float* __restrict__ abuf, const float* __restrict__ den,
                            float* out, int E, int Nn, int H, int C) {
  int e = blockIdx.x;                       // grid = Etot
  int src, dst;
  if (e < E) { src = ei[e]; dst = ei[E + e]; }
  else       { src = dst = e - E; }
  int HC = H * C;
  for (int idx = threadIdx.x; idx < HC; idx += blockDim.x) {
    int h = idx / C;
    float attn = abuf[(size_t)e * H + h] / (den[dst * H + h] + 1e-16f);
    atomicAdd(&out[(size_t)dst * HC + idx], hmat[(size_t)src * HC + idx] * attn);
  }
}

// ------------------------------------------------- bias + LayerNorm (+ReLU)

__global__ __launch_bounds__(256) void bias_ln_k(const float* __restrict__ in,
                                                 const float* __restrict__ b,
                                                 const float* __restrict__ g,
                                                 const float* __restrict__ beta,
                                                 float* out, int D, int relu) {
  __shared__ float red[256];
  int n = blockIdx.x;
  const float* ip = in + (size_t)n * D;
  float s = 0.0f;
  for (int i = threadIdx.x; i < D; i += 256) s += ip[i] + b[i];
  red[threadIdx.x] = s;
  __syncthreads();
  for (int st = 128; st > 0; st >>= 1) {
    if (threadIdx.x < st) red[threadIdx.x] += red[threadIdx.x + st];
    __syncthreads();
  }
  float mu = red[0] / (float)D;
  __syncthreads();
  float vs = 0.0f;
  for (int i = threadIdx.x; i < D; i += 256) {
    float v = ip[i] + b[i] - mu;
    vs += v * v;
  }
  red[threadIdx.x] = vs;
  __syncthreads();
  for (int st = 128; st > 0; st >>= 1) {
    if (threadIdx.x < st) red[threadIdx.x] += red[threadIdx.x + st];
    __syncthreads();
  }
  float inv = rsqrtf(red[0] / (float)D + LN_EPS);
  for (int i = threadIdx.x; i < D; i += 256) {
    float v = (ip[i] + b[i] - mu) * inv * g[i] + beta[i];
    if (relu) v = fmaxf(v, 0.0f);
    out[(size_t)n * D + i] = v;
  }
}

// ------------------------------------------------------- attention pooling

__global__ void pool_score_k(const float* __restrict__ nodes, const float* __restrict__ wa,
                             const float* __restrict__ ba, const int* __restrict__ seg,
                             float* scores, float* m, int Nn, int C) {
  int n = blockIdx.x * blockDim.x + threadIdx.x;
  if (n >= Nn) return;
  float s = 0.0f;
  for (int c = 0; c < C; c++) s += nodes[(size_t)n * C + c] * wa[c];
  s += ba[0];
  scores[n] = s;
  atomicMaxF(&m[seg[n]], s);
}

__global__ void pool_exp_k(const float* __restrict__ m, const int* __restrict__ seg,
                           float* scores, float* den, int Nn) {
  int n = blockIdx.x * blockDim.x + threadIdx.x;
  if (n >= Nn) return;
  float ex = expf(scores[n] - m[seg[n]]);
  scores[n] = ex;
  atomicAdd(&den[seg[n]], ex);
}

__global__ void pool_max_k(const float* __restrict__ nodes, const float* __restrict__ scores,
                           const float* __restrict__ den, const int* __restrict__ seg,
                           float* gpool, int Nn, int C) {
  int i = blockIdx.x * blockDim.x + threadIdx.x;
  if (i >= Nn * C) return;
  int n = i / C, c = i % C;
  float attn = scores[n] / (den[seg[n]] + 1e-16f);
  atomicMaxF(&gpool[(size_t)seg[n] * C + c], nodes[i] * attn);
}

__global__ void pool_out_k(const float* __restrict__ gpool, const float* __restrict__ aff,
                           float* out, int Bq, int C) {
  int i = blockIdx.x * blockDim.x + threadIdx.x;
  if (i >= Bq * C) return;
  int b = i / C;
  float gv = gpool[i];
  float fin = (gv - gv == 0.0f) ? gv : 0.0f;   // finite check
  out[i] = fin * (1.0f + aff[b]);
}

__global__ void aff_copy_k(const float* __restrict__ aff, float* out, int Bq) {
  int b = blockIdx.x * blockDim.x + threadIdx.x;
  if (b < Bq) out[b] = aff[b];
}

// ---------------------------------------------------------------- launcher

struct Params {
  const float *W1, *We1, *as1, *ad1, *ae1, *b1, *g1, *bn1;
  const float *W2, *We2, *as2, *ad2, *ae2, *b2, *g2, *bn2;
  const float *wa, *ba;
};

extern "C" void kernel_launch(void* const* d_in, const int* in_sizes, int n_in,
                              void* d_out, int out_size, void* d_ws, size_t ws_size,
                              hipStream_t stream) {
  (void)in_sizes; (void)n_in; (void)out_size; (void)ws_size;

  const int Nn = 30000, Ee = 90000, Bq = 512;
  const int IND = 55, H1 = 4, C1 = 256, H1C = 1024, C2 = 256;
  const int Etot = Ee + Nn;

  const float* x_d  = (const float*)d_in[0];
  const int*   ei_d = (const int*)  d_in[1];
  const float* ea_d = (const float*)d_in[2];
  const int*   sg_d = (const int*)  d_in[3];
  const float* x_f  = (const float*)d_in[4];
  const int*   ei_f = (const int*)  d_in[5];
  const float* ea_f = (const float*)d_in[6];
  const int*   sg_f = (const int*)  d_in[7];
  const float* aff  = (const float*)d_in[8];

  auto getP = [&](int base) {
    Params p;
    p.W1  = (const float*)d_in[base + 0];  p.We1 = (const float*)d_in[base + 1];
    p.as1 = (const float*)d_in[base + 2];  p.ad1 = (const float*)d_in[base + 3];
    p.ae1 = (const float*)d_in[base + 4];  p.b1  = (const float*)d_in[base + 5];
    p.g1  = (const float*)d_in[base + 6];  p.bn1 = (const float*)d_in[base + 7];
    p.W2  = (const float*)d_in[base + 8];  p.We2 = (const float*)d_in[base + 9];
    p.as2 = (const float*)d_in[base + 10]; p.ad2 = (const float*)d_in[base + 11];
    p.ae2 = (const float*)d_in[base + 12]; p.b2  = (const float*)d_in[base + 13];
    p.g2  = (const float*)d_in[base + 14]; p.bn2 = (const float*)d_in[base + 15];
    p.wa  = (const float*)d_in[base + 16]; p.ba  = (const float*)d_in[base + 17];
    return p;
  };
  Params pd = getP(9), pf = getP(27);

  // workspace bump allocator (buffers reused by both encoders, run in sequence)
  char* wsb = (char*)d_ws;
  size_t off = 0;
  auto alloc = [&](size_t elems) {
    void* p = wsb + off;
    off += (elems * sizeof(float) + 255) & ~(size_t)255;
    return (float*)p;
  };
  float* loop_attr = alloc((size_t)Nn * 6);
  float* cnt       = alloc(Nn);
  float* h1        = alloc((size_t)Nn * H1C);   // conv1 features, later LN1 output
  float* out1      = alloc((size_t)Nn * H1C);   // conv1 aggregation
  float* g2h       = alloc((size_t)Nn * C2);    // conv2 features, later final nodes
  float* out2      = alloc((size_t)Nn * C2);    // conv2 aggregation
  float* als       = alloc((size_t)Nn * H1);
  float* ald       = alloc((size_t)Nn * H1);
  float* amax      = alloc((size_t)Nn * H1);
  float* den       = alloc((size_t)Nn * H1);
  float* abuf      = alloc((size_t)Etot * H1);
  float* wea       = alloc(32);
  float* scores    = alloc(Nn);
  float* pm        = alloc(Bq);
  float* pden      = alloc(Bq);
  float* gpool     = alloc((size_t)Bq * C2);

  auto cdiv = [](int a, int b) { return (a + b - 1) / b; };
  auto fill = [&](float* p, float v, int n) {
    fill_k<<<cdiv(n, 256), 256, 0, stream>>>(p, v, n);
  };
  const int gridM = cdiv(Nn, 64);   // 64-row block tiles, M guarded

  auto encoder = [&](const float* x, const int* ei, const float* ea, const int* seg,
                     const Params& p, float* out_slice) {
    // ---- self-loop edge_attr = mean of incoming
    fill(loop_attr, 0.0f, Nn * 6);
    fill(cnt, 0.0f, Nn);
    sl_accum_k<<<cdiv(Ee, 256), 256, 0, stream>>>(ei, ea, loop_attr, cnt, Ee);
    sl_final_k<<<cdiv(Nn * 6, 256), 256, 0, stream>>>(loop_attr, cnt, Nn);

    // ---- conv1: h1 = x @ W1  (K=55 -> padded, bf16 WMMA)
    gemm_bf16_wmma_k<<<dim3(H1C / 64, gridM), 256, 0, stream>>>(x, p.W1, h1, Nn, H1C, IND);
    weal_k<<<1, 32, 0, stream>>>(p.We1, p.ae1, wea, H1, C1);
    node_alpha_k<<<cdiv(Nn * H1, 256), 256, 0, stream>>>(h1, p.as1, p.ad1, als, ald, Nn, H1, C1);
    fill(amax, -INFINITY, Nn * H1);
    fill(den, 0.0f, Nn * H1);
    edge_alpha_k<<<cdiv(Etot * H1, 256), 256, 0, stream>>>(ei, ea, loop_attr, wea, als, ald,
                                                           abuf, amax, Ee, Nn, H1);
    edge_exp_k<<<cdiv(Etot * H1, 256), 256, 0, stream>>>(ei, amax, abuf, den, Ee, Nn, H1);
    fill(out1, 0.0f, Nn * H1C);
    aggregate_k<<<Etot, 256, 0, stream>>>(ei, h1, abuf, den, out1, Ee, Nn, H1, C1);
    // bias + LN + ReLU -> h1 (reuse)
    bias_ln_k<<<Nn, 256, 0, stream>>>(out1, p.b1, p.g1, p.bn1, h1, H1C, 1);

    // ---- conv2: g2h = h1 @ W2 (K=1024, bf16 WMMA)
    gemm_bf16_wmma_k<<<dim3(C2 / 64, gridM), 256, 0, stream>>>(h1, p.W2, g2h, Nn, C2, H1C);
    weal_k<<<1, 32, 0, stream>>>(p.We2, p.ae2, wea, 1, C2);
    node_alpha_k<<<cdiv(Nn, 256), 256, 0, stream>>>(g2h, p.as2, p.ad2, als, ald, Nn, 1, C2);
    fill(amax, -INFINITY, Nn);
    fill(den, 0.0f, Nn);
    edge_alpha_k<<<cdiv(Etot, 256), 256, 0, stream>>>(ei, ea, loop_attr, wea, als, ald,
                                                      abuf, amax, Ee, Nn, 1);
    edge_exp_k<<<cdiv(Etot, 256), 256, 0, stream>>>(ei, amax, abuf, den, Ee, Nn, 1);
    fill(out2, 0.0f, Nn * C2);
    aggregate_k<<<Etot, 256, 0, stream>>>(ei, g2h, abuf, den, out2, Ee, Nn, 1, C2);
    // bias + LN -> nodes (reuse g2h)
    bias_ln_k<<<Nn, 256, 0, stream>>>(out2, p.b2, p.g2, p.bn2, g2h, C2, 0);

    // ---- attention pooling + scale
    fill(pm, -INFINITY, Bq);
    fill(pden, 0.0f, Bq);
    fill(gpool, -INFINITY, Bq * C2);
    pool_score_k<<<cdiv(Nn, 256), 256, 0, stream>>>(g2h, p.wa, p.ba, seg, scores, pm, Nn, C2);
    pool_exp_k<<<cdiv(Nn, 256), 256, 0, stream>>>(pm, seg, scores, pden, Nn);
    pool_max_k<<<cdiv(Nn * C2, 256), 256, 0, stream>>>(g2h, scores, pden, seg, gpool, Nn, C2);
    pool_out_k<<<cdiv(Bq * C2, 256), 256, 0, stream>>>(gpool, aff, out_slice, Bq, C2);
  };

  float* out = (float*)d_out;
  encoder(x_d, ei_d, ea_d, sg_d, pd, out);                       // d_fea * scale
  encoder(x_f, ei_f, ea_f, sg_f, pf, out + (size_t)Bq * C2);     // f_fea * scale
  aff_copy_k<<<cdiv(Bq, 256), 256, 0, stream>>>(aff, out + (size_t)2 * Bq * C2, Bq);
}
